// STAttentionPooling_34230889349559
// MI455X (gfx1250) — compile-verified, matrix operand order checked by reading, not run on hardware
//
#include <hip/hip_runtime.h>
#include <hip/hip_bf16.h>
#include <math.h>

#define BB   8
#define NNN  1024
#define TT   16
#define EE   512
#define HH   8
#define DHH  64
#define KNN_K 16

typedef __attribute__((ext_vector_type(8)))  float  v8f;
typedef __attribute__((ext_vector_type(16))) __bf16 v16bf;
typedef __attribute__((ext_vector_type(8)))  __bf16 v8bf;

#define WMMA_BF16(A, B, C) \
  __builtin_amdgcn_wmma_f32_16x16x32_bf16(false, (A), false, (B), (short)0, (C), false, false)

// ---------------------------------------------------------------------------
// bf16 split helpers (integer-domain RNE)
// ---------------------------------------------------------------------------
__device__ __forceinline__ unsigned short f2bf(float x) {
  unsigned u = __builtin_bit_cast(unsigned, x);
  unsigned r = (u + 0x7FFFu + ((u >> 16) & 1u)) >> 16;
  return (unsigned short)r;
}
__device__ __forceinline__ float bf2f(unsigned short b) {
  return __builtin_bit_cast(float, (unsigned)b << 16);
}
// split a float4 into hi/lo bf16 quads stored at float4-index idx4
__device__ __forceinline__ void split_store_u2(__bf16* hi, __bf16* lo, size_t idx4, float4 f) {
  unsigned short h0 = f2bf(f.x), h1 = f2bf(f.y), h2 = f2bf(f.z), h3 = f2bf(f.w);
  unsigned short r0 = f2bf(f.x - bf2f(h0)), r1 = f2bf(f.y - bf2f(h1));
  unsigned short r2 = f2bf(f.z - bf2f(h2)), r3 = f2bf(f.w - bf2f(h3));
  ((uint2*)hi)[idx4] = make_uint2(((unsigned)h1 << 16) | h0, ((unsigned)h3 << 16) | h2);
  ((uint2*)lo)[idx4] = make_uint2(((unsigned)r1 << 16) | r0, ((unsigned)r3 << 16) | r2);
}

// ---------------------------------------------------------------------------
// CDNA5 async copy: global -> LDS (16B per lane), tracked by ASYNCcnt
// ---------------------------------------------------------------------------
__device__ __forceinline__ void async_ld128(unsigned lds_addr, const void* sbase, unsigned voff) {
  asm volatile("global_load_async_to_lds_b128 %0, %1, %2"
               :: "v"(lds_addr), "v"(voff), "s"(sbase)
               : "memory");
}
__device__ __forceinline__ void wait_async0() {
#if defined(__has_builtin)
#if __has_builtin(__builtin_amdgcn_s_wait_asynccnt)
  __builtin_amdgcn_s_wait_asynccnt(0);
#else
  asm volatile("s_wait_asynccnt 0x0" ::: "memory");
#endif
#else
  asm volatile("s_wait_asynccnt 0x0" ::: "memory");
#endif
}

// ---------------------------------------------------------------------------
// Kernel 1: exact-KNN over 1024 nodes + distance-bias weights
// ---------------------------------------------------------------------------
__global__ __launch_bounds__(256) void knn_kernel(
    const float* __restrict__ pos, const float* __restrict__ tau_p,
    int* __restrict__ knn_idx, float* __restrict__ log_w)
{
  __shared__ float sx[NNN], sy[NNN], sq[NNN];
  const int tid = threadIdx.x;
  for (int j = tid; j < NNN; j += 256) {
    float x = pos[2 * j], y = pos[2 * j + 1];
    sx[j] = x; sy[j] = y; sq[j] = x * x + y * y;
  }
  __syncthreads();

  const int i = blockIdx.x * 256 + tid;
  const float xi = sx[i], yi = sy[i], qi = sq[i];

  float bd[KNN_K];
  int   bi[KNN_K];
#pragma unroll
  for (int t = 0; t < KNN_K; ++t) { bd[t] = __builtin_inff(); bi[t] = 0; }

  for (int j = 0; j < NNN; ++j) {
    float d2 = qi + sq[j] - 2.0f * (xi * sx[j] + yi * sy[j]);
    d2 = fmaxf(d2, 0.0f);
    float d = sqrtf(d2 + 1e-9f);
    if (j == i) d = __builtin_inff();
    if (d < bd[KNN_K - 1]) {
      int p = KNN_K - 1;
      while (p > 0 && bd[p - 1] > d) { bd[p] = bd[p - 1]; bi[p] = bi[p - 1]; --p; }
      bd[p] = d; bi[p] = j;
    }
  }

  float mean = 0.0f;
#pragma unroll
  for (int t = 0; t < KNN_K; ++t) mean += bd[t];
  mean *= (1.0f / (float)KNN_K);
  const float scale    = fmaxf(mean, 1.1920929e-7f);   // FLT_EPSILON
  const float tau_safe = fmaxf(tau_p[0], 1e-4f);       // TAU_MIN
#pragma unroll
  for (int t = 0; t < KNN_K; ++t) {
    float lw = fminf(fmaxf(-(bd[t] / scale) / tau_safe, -10.0f), 0.0f);
    knn_idx[i * KNN_K + t] = bi[t];
    log_w[i * KNN_K + t]   = lw;
  }
}

// ---------------------------------------------------------------------------
// Kernel 2: fused tail reduce + bf16 split planes
//   Hhi/Hlo <- split(hidden[:,:,T-1,:]);  Mhi/Mlo <- split(mean last-4)
// ---------------------------------------------------------------------------
__global__ __launch_bounds__(256) void tail_reduce_split_kernel(
    const float* __restrict__ hidden,
    __bf16* __restrict__ Hhi, __bf16* __restrict__ Hlo,
    __bf16* __restrict__ Mhi, __bf16* __restrict__ Mlo)
{
  const int E4 = EE / 4;
  const int i  = blockIdx.x * 256 + threadIdx.x;      // over BB*NNN*E4 exactly
  const int bn = i / E4;
  const int e4 = i - bn * E4;
  const float4* src = (const float4*)hidden + (size_t)bn * TT * E4 + (size_t)(TT - 4) * E4 + e4;
  float4 t0 = src[0], t1 = src[E4], t2 = src[2 * E4], t3 = src[3 * E4];
  float4 m;
  m.x = (t0.x + t1.x + t2.x + t3.x) * 0.25f;
  m.y = (t0.y + t1.y + t2.y + t3.y) * 0.25f;
  m.z = (t0.z + t1.z + t2.z + t3.z) * 0.25f;
  m.w = (t0.w + t1.w + t2.w + t3.w) * 0.25f;
  const size_t o = (size_t)bn * E4 + e4;
  split_store_u2(Hhi, Hlo, o, t3);
  split_store_u2(Mhi, Mlo, o, m);
}

// ---------------------------------------------------------------------------
// Kernel 3: generic fp32 -> bf16 hi/lo plane split (for weight matrices)
// ---------------------------------------------------------------------------
__global__ __launch_bounds__(256) void split_planes_kernel(
    const float* __restrict__ in, __bf16* __restrict__ hi, __bf16* __restrict__ lo, int n4)
{
  const int i = blockIdx.x * 256 + threadIdx.x;
  if (i >= n4) return;
  split_store_u2(hi, lo, (size_t)i, ((const float4*)in)[i]);
}

// ---------------------------------------------------------------------------
// Kernel 4: Y[M,N] = X @ W^T + bias with split-bf16 x3 WMMA, async LDS fills
// ---------------------------------------------------------------------------
#define BM   64
#define BN   64
#define BKK  32
#define LSTR 40   // bf16 elems per LDS row (32 + pad; 80B rows keep 16B chunks)

// A/B fragment: lane holds row (l%16); K-chunks [half*8, half*8+8) and [16+half*8, ...)
__device__ __forceinline__ v16bf frag_load(const __bf16* rowp, int half) {
  v16bf r;
  *(v8bf*)&r         = *(const v8bf*)(rowp + half * 8);
  *(((v8bf*)&r) + 1) = *(const v8bf*)(rowp + 16 + half * 8);
  return r;
}

__global__ __launch_bounds__(256) void gemm_bf16x3_kernel(
    const __bf16* __restrict__ Xhi, const __bf16* __restrict__ Xlo,
    const __bf16* __restrict__ Whi, const __bf16* __restrict__ Wlo,
    const float* __restrict__ bias, float* __restrict__ Y,
    int M, int Ncols, int K)
{
  __shared__ __align__(16) __bf16 Ahi[BM * LSTR];
  __shared__ __align__(16) __bf16 Alo[BM * LSTR];
  __shared__ __align__(16) __bf16 Bhi[BN * LSTR];
  __shared__ __align__(16) __bf16 Blo[BN * LSTR];

  const int tid  = threadIdx.x;
  const int lane = tid & 31;
  const int half = lane >> 4;
  const int l16  = lane & 15;
  const int wave = tid >> 5;
  const int mw   = (wave & 3) * 16;   // wave's M subtile within block
  const int nw   = (wave >> 2) * 32;  // wave's N 32-col group (2 subtiles)
  const int m0   = blockIdx.x * BM;
  const int n0   = blockIdx.y * BN;

  // async fill mapping: 256 threads x 16B = 4KB per plane (64 rows x 64B)
  const int row   = tid >> 2;          // 0..63
  const int chunk = tid & 3;           // 0..3 (16B chunks within a row)
  const unsigned ldsOff = (unsigned)(row * (LSTR * 2) + chunk * 16);
  // generic->32-bit truncation of a shared pointer yields the LDS byte offset
  const unsigned aAhi = (unsigned)(unsigned long long)(void*)Ahi + ldsOff;
  const unsigned aAlo = (unsigned)(unsigned long long)(void*)Alo + ldsOff;
  const unsigned aBhi = (unsigned)(unsigned long long)(void*)Bhi + ldsOff;
  const unsigned aBlo = (unsigned)(unsigned long long)(void*)Blo + ldsOff;
  const unsigned offA0 = (unsigned)(((m0 + row) * K + chunk * 8) * 2);  // bytes
  const unsigned offB0 = (unsigned)(((n0 + row) * K + chunk * 8) * 2);  // bytes

  v8f acc0 = {};
  v8f acc1 = {};

  for (int kb = 0; kb < K; kb += BKK) {
    const unsigned kOff = (unsigned)(kb * 2);
    __syncthreads();                       // prior reads of LDS complete
    async_ld128(aAhi, Xhi, offA0 + kOff);
    async_ld128(aAlo, Xlo, offA0 + kOff);
    async_ld128(aBhi, Whi, offB0 + kOff);
    async_ld128(aBlo, Wlo, offB0 + kOff);
    wait_async0();                         // this wave's fills landed
    __syncthreads();                       // all waves' fills landed

    const __bf16* pa  = Ahi + (mw + l16) * LSTR;
    const __bf16* pal = Alo + (mw + l16) * LSTR;
    v16bf a1 = frag_load(pa,  half);
    v16bf a2 = frag_load(pal, half);

    const __bf16* pb  = Bhi + (nw + l16) * LSTR;
    const __bf16* pbl = Blo + (nw + l16) * LSTR;
    v16bf b1 = frag_load(pb,  half);
    v16bf b2 = frag_load(pbl, half);
    acc0 = WMMA_BF16(a1, b1, acc0);
    acc0 = WMMA_BF16(a2, b1, acc0);
    acc0 = WMMA_BF16(a1, b2, acc0);

    v16bf c1 = frag_load(pb  + 16 * LSTR, half);
    v16bf c2 = frag_load(pbl + 16 * LSTR, half);
    acc1 = WMMA_BF16(a1, c1, acc1);
    acc1 = WMMA_BF16(a2, c1, acc1);
    acc1 = WMMA_BF16(a1, c2, acc1);
  }

  const int col0 = n0 + nw + l16;
  const float bz0 = bias[col0];
  const float bz1 = bias[col0 + 16];
#pragma unroll
  for (int v = 0; v < 8; ++v) {
    const int rowy = m0 + mw + v + 8 * half;    // C/D layout: VGPR v -> M = v + 8*half
    Y[(size_t)rowy * Ncols + col0]      = acc0[v] + bz0;
    Y[(size_t)rowy * Ncols + col0 + 16] = acc1[v] + bz1;
  }
}

// ---------------------------------------------------------------------------
// Kernel 5: per-(b,n,h) 16-neighbor attention; emits bf16 split ctx planes
// ---------------------------------------------------------------------------
__global__ __launch_bounds__(256) void attn_kernel(
    const float* __restrict__ Q, const float* __restrict__ Kb,
    const float* __restrict__ Vb, const int* __restrict__ knn_idx,
    const float* __restrict__ log_w,
    __bf16* __restrict__ Chi, __bf16* __restrict__ Clo)
{
  const int gt   = blockIdx.x * 256 + threadIdx.x;
  const int lane = gt & 31;
  const int wid  = gt >> 5;            // (b*NNN + n)*HH + h
  const int h    = wid & (HH - 1);
  const int bn   = wid >> 3;           // b*NNN + n
  const int n    = bn & (NNN - 1);
  const size_t rowOff = (size_t)bn * EE + h * DHH;

  const float2 qv = ((const float2*)(Q + rowOff))[lane];   // dims 2*lane, 2*lane+1

  int nbr[KNN_K];
  const int* idx = knn_idx + n * KNN_K;
#pragma unroll
  for (int j = 0; j < KNN_K; ++j) nbr[j] = idx[j];

  const float* lwp = log_w + n * KNN_K;
  float logits[KNN_K];
#pragma unroll
  for (int j = 0; j < KNN_K; ++j) {
    const size_t kOff = (size_t)(bn - n + nbr[j]) * EE + h * DHH;  // (b*NNN+nbr)*EE
    const float2 kv = ((const float2*)(Kb + kOff))[lane];
    float p = qv.x * kv.x + qv.y * kv.y;
#pragma unroll
    for (int off = 16; off >= 1; off >>= 1) p += __shfl_xor(p, off, 32);
    logits[j] = p * 0.125f + lwp[j];   // 1/sqrt(64) = 0.125
  }

  float mx = logits[0];
#pragma unroll
  for (int j = 1; j < KNN_K; ++j) mx = fmaxf(mx, logits[j]);
  float w[KNN_K];
  float s = 0.0f;
#pragma unroll
  for (int j = 0; j < KNN_K; ++j) { w[j] = __expf(logits[j] - mx); s += w[j]; }
  const float inv = 1.0f / s;

  float cx = 0.0f, cy = 0.0f;
#pragma unroll
  for (int j = 0; j < KNN_K; ++j) {
    const size_t vOff = (size_t)(bn - n + nbr[j]) * EE + h * DHH;
    const float2 vv = ((const float2*)(Vb + vOff))[lane];
    const float wj = w[j] * inv;
    cx = fmaf(wj, vv.x, cx);
    cy = fmaf(wj, vv.y, cy);
  }

  const unsigned short hx = f2bf(cx), hy = f2bf(cy);
  const unsigned short rx = f2bf(cx - bf2f(hx)), ry = f2bf(cy - bf2f(hy));
  ((unsigned*)Chi)[(rowOff >> 1) + lane] = ((unsigned)hy << 16) | hx;
  ((unsigned*)Clo)[(rowOff >> 1) + lane] = ((unsigned)ry << 16) | rx;
}

// ---------------------------------------------------------------------------
// Launch
// ---------------------------------------------------------------------------
extern "C" void kernel_launch(void* const* d_in, const int* in_sizes, int n_in,
                              void* d_out, int out_size, void* d_ws, size_t ws_size,
                              hipStream_t stream)
{
  const float* hidden = (const float*)d_in[0];
  const float* pos    = (const float*)d_in[1];
  const float* Wq = (const float*)d_in[2];
  const float* bq = (const float*)d_in[3];
  const float* Wk = (const float*)d_in[4];
  const float* bk = (const float*)d_in[5];
  const float* Wv = (const float*)d_in[6];
  const float* bv = (const float*)d_in[7];
  const float* Wo = (const float*)d_in[8];
  const float* bo = (const float*)d_in[9];
  const float* tau = (const float*)d_in[10];
  float* out = (float*)d_out;
  float* ws  = (float*)d_ws;

  const size_t CH  = (size_t)BB * NNN * EE;   // 4,194,304 elems per (B,N,E) plane
  const size_t WSZ = (size_t)EE * EE;         // 262,144 elems per weight matrix

  float* Qm = ws;
  float* Km = ws + CH;
  float* Vm = ws + 2 * CH;
  __bf16* bfb = (__bf16*)(ws + 3 * CH);
  __bf16* Hhi = bfb + 0 * CH;
  __bf16* Hlo = bfb + 1 * CH;
  __bf16* Mhi = bfb + 2 * CH;
  __bf16* Mlo = bfb + 3 * CH;
  __bf16* Chi = bfb + 4 * CH;
  __bf16* Clo = bfb + 5 * CH;
  __bf16* Wqhi = bfb + 6 * CH;
  __bf16* Wqlo = Wqhi + WSZ;
  __bf16* Wkhi = Wqhi + 2 * WSZ;
  __bf16* Wklo = Wqhi + 3 * WSZ;
  __bf16* Wvhi = Wqhi + 4 * WSZ;
  __bf16* Wvlo = Wqhi + 5 * WSZ;
  __bf16* Wohi = Wqhi + 6 * WSZ;
  __bf16* Wolo = Wqhi + 7 * WSZ;
  int*   knn = (int*)(Wqhi + 8 * WSZ);
  float* lw  = (float*)(knn + (size_t)NNN * KNN_K);

  knn_kernel<<<NNN / 256, 256, 0, stream>>>(pos, tau, knn, lw);
  tail_reduce_split_kernel<<<(BB * NNN * (EE / 4)) / 256, 256, 0, stream>>>(
      hidden, Hhi, Hlo, Mhi, Mlo);

  const int w4 = (int)(WSZ / 4);
  split_planes_kernel<<<w4 / 256, 256, 0, stream>>>(Wq, Wqhi, Wqlo, w4);
  split_planes_kernel<<<w4 / 256, 256, 0, stream>>>(Wk, Wkhi, Wklo, w4);
  split_planes_kernel<<<w4 / 256, 256, 0, stream>>>(Wv, Wvhi, Wvlo, w4);
  split_planes_kernel<<<w4 / 256, 256, 0, stream>>>(Wo, Wohi, Wolo, w4);

  dim3 gg(BB * NNN / BM, EE / BN);            // 128 x 8
  gemm_bf16x3_kernel<<<gg, 256, 0, stream>>>(Hhi, Hlo, Wqhi, Wqlo, bq, Qm, BB * NNN, EE, EE);
  gemm_bf16x3_kernel<<<gg, 256, 0, stream>>>(Mhi, Mlo, Wkhi, Wklo, bk, Km, BB * NNN, EE, EE);
  gemm_bf16x3_kernel<<<gg, 256, 0, stream>>>(Mhi, Mlo, Wvhi, Wvlo, bv, Vm, BB * NNN, EE, EE);

  attn_kernel<<<(BB * NNN * HH * 32) / 256, 256, 0, stream>>>(Qm, Km, Vm, knn, lw, Chi, Clo);

  gemm_bf16x3_kernel<<<gg, 256, 0, stream>>>(Chi, Clo, Wohi, Wolo, bo, out, BB * NNN, EE, EE);

  (void)in_sizes; (void)n_in; (void)out_size; (void)ws_size;
}
